// BailingMoE_67748814127135
// MI455X (gfx1250) — matrix operand order repaired and checked by reading, not compiled, not run on hardware
//
#include <hip/hip_runtime.h>
#include <hip/hip_bf16.h>

#define H_DIM 2048
#define E_NUM 32
#define I_DIM 1408
#define TOPK 6
#define IS_DIM 2816
#define T_NUM 2048

typedef __bf16 v16bf __attribute__((ext_vector_type(16)));
typedef float  v8f   __attribute__((ext_vector_type(8)));
typedef unsigned int v8u   __attribute__((ext_vector_type(8)));
typedef unsigned int u32x4 __attribute__((ext_vector_type(4)));
typedef int          i32x8 __attribute__((ext_vector_type(8)));
typedef int          i32x4 __attribute__((ext_vector_type(4)));

__device__ __forceinline__ unsigned short f32_to_bf16(float f) {
    unsigned int u = __float_as_uint(f);
    u += 0x7FFFu + ((u >> 16) & 1u);   // round-to-nearest-even
    return (unsigned short)(u >> 16);
}

// ---------------------------------------------------------------------------
// TDM: async 2D tile load (bf16) global -> LDS, padded LDS rows.
// tile = 128 rows x 32 bf16 (64B); LDS row stride 80B via pad feature
// (pad_interval: 16 DWORDs -> code 3, pad_amount: 4 DWORDs -> code 3).
// rowsRemain clips dim1 so OOB rows zero-fill (partial M tiles).
// ---------------------------------------------------------------------------
__device__ __forceinline__ void tdm_load_tile_a(const unsigned short* gbase,
                                                unsigned ldsOff,
                                                int rowsRemain, int lda) {
    unsigned long long ga = (unsigned long long)(size_t)gbase;
    u32x4 g0;
    g0[0] = 1u;                                            // count=1, no gather
    g0[1] = ldsOff;                                        // lds_addr [63:32]
    g0[2] = (unsigned)(ga & 0xFFFFFFFFu);                  // global_addr lo
    g0[3] = (unsigned)((ga >> 32) & 0x01FFFFFFu) | (2u << 30); // addr hi + type=2

    i32x8 g1;
    g1[0] = (1 << 16) | (1 << 20) | (3 << 22) | (3 << 25); // 2B, pad en, 16dw/4dw
    unsigned td0 = 0x7FFFFFFFu;                            // dim0 never OOB
    unsigned td1 = (unsigned)rowsRemain;                   // clip rows
    g1[1] = (int)((td0 & 0xFFFFu) << 16);                  // tensor_dim0 lo16
    g1[2] = (int)((td0 >> 16) | ((td1 & 0xFFFFu) << 16));  // dim0 hi / dim1 lo
    g1[3] = (int)((td1 >> 16) | (32u << 16));              // dim1 hi / tile_dim0=32
    g1[4] = 128;                                           // tile_dim1=128, dim2=0
    unsigned long long s0 = (unsigned long long)lda;       // dim0 stride (elems)
    g1[5] = (int)(s0 & 0xFFFFFFFFu);
    g1[6] = (int)((s0 >> 32) & 0xFFFFu);                   // stride hi / dim1_stride=0
    g1[7] = 0;
    i32x4 gz4 = {0, 0, 0, 0};
    i32x8 gz8 = {0, 0, 0, 0, 0, 0, 0, 0};
    __builtin_amdgcn_tensor_load_to_lds(g0, g1, gz4, gz4, gz8, 0);
}

// ---------------------------------------------------------------------------
// Small helper kernels
// ---------------------------------------------------------------------------
__global__ void init_counts_kernel(int* counts) {
    if (threadIdx.x < E_NUM) counts[threadIdx.x] = 0;
}

__global__ void cvt_bf16_kernel(const float* __restrict__ x,
                                unsigned short* __restrict__ xb, int n) {
    int i = (blockIdx.x * blockDim.x + threadIdx.x) * 4;
    if (i < n) {
        float4 v = *(const float4*)(x + i);
        xb[i + 0] = f32_to_bf16(v.x);
        xb[i + 1] = f32_to_bf16(v.y);
        xb[i + 2] = f32_to_bf16(v.z);
        xb[i + 3] = f32_to_bf16(v.w);
    }
}

// One wave (32 lanes) per token; lane == expert.
__global__ __launch_bounds__(32)
void router_kernel(const float* __restrict__ x, const float* __restrict__ gate_w,
                   int* __restrict__ topk_idx, float* __restrict__ topk_w,
                   int* __restrict__ counts) {
    const int t = blockIdx.x;
    const int lane = threadIdx.x;

    const float4* xr = (const float4*)(x + (size_t)t * H_DIM);
    const float4* wr = (const float4*)(gate_w + (size_t)lane * H_DIM);
    float acc = 0.f;
    for (int k = 0; k < H_DIM / 4; ++k) {
        float4 a = xr[k];
        float4 b = wr[k];
        acc += a.x * b.x + a.y * b.y + a.z * b.z + a.w * b.w;
    }
    // softmax over the 32 experts (one per lane)
    float m = acc;
    for (int off = 16; off > 0; off >>= 1) m = fmaxf(m, __shfl_xor(m, off, 32));
    float p = __expf(acc - m);
    float s = p;
    for (int off = 16; off > 0; off >>= 1) s += __shfl_xor(s, off, 32);
    float prob = p / s;

    // top-6 with deterministic tie-break, then renormalize
    float myp = prob;
    int   seli[TOPK];
    float selv[TOPK];
    float wsum = 0.f;
    for (int j = 0; j < TOPK; ++j) {
        float v = myp;
        int   idx = lane;
        for (int off = 16; off > 0; off >>= 1) {
            float ov = __shfl_xor(v, off, 32);
            int   oi = __shfl_xor(idx, off, 32);
            if (ov > v || (ov == v && oi < idx)) { v = ov; idx = oi; }
        }
        seli[j] = idx;
        selv[j] = v;
        wsum += v;
        if (lane == idx) myp = -1.0f;
    }
    if (lane < TOPK) {
        int e = seli[lane];
        topk_idx[t * TOPK + lane] = e;
        topk_w[t * TOPK + lane]   = selv[lane] / wsum;
        atomicAdd(&counts[e], 1);
    }
}

__global__ void scan_kernel(const int* __restrict__ counts,
                            int* __restrict__ offsets, int* __restrict__ cursors) {
    if (threadIdx.x == 0) {
        int acc = 0;
        for (int e = 0; e < E_NUM; ++e) {
            offsets[e] = acc;
            cursors[e] = acc;
            acc += counts[e];
        }
    }
}

__global__ void scatter_kernel(const int* __restrict__ topk_idx,
                               const float* __restrict__ topk_w,
                               int* __restrict__ cursors,
                               int* __restrict__ rowToken,
                               float* __restrict__ rowWeight) {
    int i = blockIdx.x * blockDim.x + threadIdx.x;
    if (i < T_NUM * TOPK) {
        int e = topk_idx[i];
        int pos = atomicAdd(&cursors[e], 1);
        rowToken[pos]  = i / TOPK;
        rowWeight[pos] = topk_w[i];
    }
}

// ---------------------------------------------------------------------------
// Tiled WMMA GEMM. MODE: 0 = SwiGLU(gate,up) -> bf16 act out
//                        1 = plain f32 store
//                        2 = per-row scale + atomic accumulate into out
// GATHER: A rows gathered through rowToken (expert gate_up pass).
// Block 256 threads = 8 waves; tile 128(M) x 64(N); K step 32 (bf16 WMMA).
// Non-gather passes fetch the A tile with the Tensor Data Mover (async DMA,
// TENSORcnt), overlapped with VALU f32->bf16 conversion of the B tiles.
// ---------------------------------------------------------------------------
template <int MODE, bool GATHER>
__global__ __launch_bounds__(256)
void gemm_wmma_kernel(const unsigned short* __restrict__ Abase, int lda,
                      const float* __restrict__ Wbase, long long wStride, int ldw,
                      int upOff,
                      void* __restrict__ Out, int ldo,
                      int Ntot, int Ktot,
                      const int* __restrict__ counts, const int* __restrict__ offsets,
                      int fullM,
                      const int* __restrict__ rowToken,
                      const float* __restrict__ rowWeight) {
    const int e = blockIdx.z;
    int M, mbase;
    if (counts) { M = counts[e]; mbase = offsets[e]; }
    else        { M = fullM;     mbase = 0; }

    const int mt = blockIdx.y * 128;
    if (mt >= M) return;                      // block-uniform early exit
    const int nt = blockIdx.x * 64;
    const float* Wm = Wbase + (long long)e * wStride;

    // padded strides (40 shorts = 80 B, multiple of 16 B -> b128-aligned)
    __shared__ __align__(16) unsigned short As[128 * 40];
    __shared__ __align__(16) unsigned short Bg[64 * 40];
    __shared__ __align__(16) unsigned short Bu[64 * 40];

    const int tid   = threadIdx.x;
    const int wave  = tid >> 5;               // 0..7 : M sub-tile
    const int lane  = tid & 31;
    const int lrow  = lane & 15;
    const int lhalf = lane >> 4;

    v8f accg[4], accu[4];
    for (int i = 0; i < 4; ++i) { accg[i] = (v8f){}; accu[i] = (v8f){}; }

    // A tile loader indices (manual/gather path): 2 threads/row, 8 bf16 (one uint4) each
    const int arow  = tid >> 1;
    const int ahalf = tid & 1;
    int aglob = -1;
    if (GATHER && (mt + arow < M)) aglob = rowToken[mbase + mt + arow];
    // B tile loader indices: 4 threads/row, 8 f32 each
    const int brow = tid >> 2;
    const int bq   = tid & 3;

    const unsigned asLds = (unsigned)(size_t)(void*)As;
    const unsigned short* aTileBase =
        GATHER ? nullptr : (Abase + (size_t)(mbase + mt) * lda);

    for (int k0 = 0; k0 < Ktot; k0 += 32) {
        __syncthreads();
        // ---- A tile -----------------------------------------------------
        if (!GATHER) {
            if (wave == 0 && lane == 0)
                tdm_load_tile_a(aTileBase + k0, asLds, M - mt, lda);
        } else {
            uint4 v = make_uint4(0u, 0u, 0u, 0u);
            if (aglob >= 0)
                v = *(const uint4*)(Abase + (size_t)aglob * lda + k0 + ahalf * 16);
            *(uint4*)(&As[arow * 40 + ahalf * 16]) = v;
        }
        // ---- B gate tile: f32 -> bf16 ----------------------------------
        {
            const float* src = Wm + (size_t)(nt + brow) * ldw + k0 + bq * 8;
            float4 f0 = *(const float4*)(src);
            float4 f1 = *(const float4*)(src + 4);
            unsigned short* dst = &Bg[brow * 40 + bq * 8];
            dst[0] = f32_to_bf16(f0.x); dst[1] = f32_to_bf16(f0.y);
            dst[2] = f32_to_bf16(f0.z); dst[3] = f32_to_bf16(f0.w);
            dst[4] = f32_to_bf16(f1.x); dst[5] = f32_to_bf16(f1.y);
            dst[6] = f32_to_bf16(f1.z); dst[7] = f32_to_bf16(f1.w);
            if (k0 + 32 < Ktot) __builtin_prefetch(src + 32, 0, 1);
        }
        // ---- B up tile (SwiGLU mode only) ------------------------------
        if (MODE == 0) {
            const float* src = Wm + (size_t)(upOff + nt + brow) * ldw + k0 + bq * 8;
            float4 f0 = *(const float4*)(src);
            float4 f1 = *(const float4*)(src + 4);
            unsigned short* dst = &Bu[brow * 40 + bq * 8];
            dst[0] = f32_to_bf16(f0.x); dst[1] = f32_to_bf16(f0.y);
            dst[2] = f32_to_bf16(f0.z); dst[3] = f32_to_bf16(f0.w);
            dst[4] = f32_to_bf16(f1.x); dst[5] = f32_to_bf16(f1.y);
            dst[6] = f32_to_bf16(f1.z); dst[7] = f32_to_bf16(f1.w);
            if (k0 + 32 < Ktot) __builtin_prefetch(src + 32, 0, 1);
        }
        if (!GATHER && wave == 0)
            __builtin_amdgcn_s_wait_tensorcnt(0);     // TDM tile landed
        __syncthreads();

        // ---- A fragment: 16x32 bf16, contiguous k order (k = 16*half + s) -----------
        // lane<16: K 0-15 ; lane>=16: K 16-31
        v8u au;
        {
            const unsigned short* ap = &As[(wave * 16 + lrow) * 40 + lhalf * 16];
            uint4 lo = *(const uint4*)(ap);
            uint4 hi = *(const uint4*)(ap + 8);
            au[0] = lo.x; au[1] = lo.y; au[2] = lo.z; au[3] = lo.w;
            au[4] = hi.x; au[5] = hi.y; au[6] = hi.z; au[7] = hi.w;
        }
        v16bf af = __builtin_bit_cast(v16bf, au);

        // ---- preload ALL B fragments, then dense WMMA issue ------------
        v16bf bfr[4], ufr[4];
#pragma unroll
        for (int ns = 0; ns < 4; ++ns) {
            const unsigned short* bp = &Bg[(ns * 16 + lrow) * 40 + lhalf * 16];
            uint4 lo = *(const uint4*)(bp);
            uint4 hi = *(const uint4*)(bp + 8);
            v8u bu; bu[0] = lo.x; bu[1] = lo.y; bu[2] = lo.z; bu[3] = lo.w;
                    bu[4] = hi.x; bu[5] = hi.y; bu[6] = hi.z; bu[7] = hi.w;
            bfr[ns] = __builtin_bit_cast(v16bf, bu);
        }
        if (MODE == 0) {
#pragma unroll
            for (int ns = 0; ns < 4; ++ns) {
                const unsigned short* up = &Bu[(ns * 16 + lrow) * 40 + lhalf * 16];
                uint4 lo = *(const uint4*)(up);
                uint4 hi = *(const uint4*)(up + 8);
                v8u uu; uu[0] = lo.x; uu[1] = lo.y; uu[2] = lo.z; uu[3] = lo.w;
                        uu[4] = hi.x; uu[5] = hi.y; uu[6] = hi.z; uu[7] = hi.w;
                ufr[ns] = __builtin_bit_cast(v16bf, uu);
            }
        }
#pragma unroll
        for (int ns = 0; ns < 4; ++ns)
            accg[ns] = __builtin_amdgcn_wmma_f32_16x16x32_bf16(
                false, af, false, bfr[ns], (short)0, accg[ns], false, false);
        if (MODE == 0) {
#pragma unroll
            for (int ns = 0; ns < 4; ++ns)
                accu[ns] = __builtin_amdgcn_wmma_f32_16x16x32_bf16(
                    false, af, false, ufr[ns], (short)0, accu[ns], false, false);
        }
    }

    // ---- epilogue: D layout lane<16 -> M=v, lane>=16 -> M=v+8 ----------
    for (int ns = 0; ns < 4; ++ns) {
        const int ncol = nt + ns * 16 + lrow;
        for (int v = 0; v < 8; ++v) {
            int ml = wave * 16 + v + 8 * lhalf;
            int gr = mt + ml;
            if (gr >= M) continue;
            if (MODE == 0) {
                float g = accg[ns][v], u = accu[ns][v];
                float hval = (g / (1.0f + __expf(-g))) * u;   // silu(g)*u
                ((unsigned short*)Out)[(size_t)(mbase + gr) * ldo + ncol] =
                    f32_to_bf16(hval);
            } else if (MODE == 1) {
                ((float*)Out)[(size_t)gr * ldo + ncol] = accg[ns][v];
            } else {
                int   ridx = mbase + gr;
                int   tok  = rowToken[ridx];
                float w    = rowWeight[ridx];
                atomicAdd(&((float*)Out)[(size_t)tok * ldo + ncol],
                          accg[ns][v] * w);
            }
        }
    }
}

// ---------------------------------------------------------------------------
extern "C" void kernel_launch(void* const* d_in, const int* in_sizes, int n_in,
                              void* d_out, int out_size, void* d_ws, size_t ws_size,
                              hipStream_t stream) {
    const float* x      = (const float*)d_in[0];
    const float* gate_w = (const float*)d_in[1];
    const float* w1     = (const float*)d_in[2];
    const float* w2     = (const float*)d_in[3];
    const float* ws1    = (const float*)d_in[4];
    const float* ws2    = (const float*)d_in[5];
    float* out = (float*)d_out;
    (void)in_sizes; (void)n_in; (void)out_size; (void)ws_size;

    char* p = (char*)d_ws;
    auto alloc = [&](size_t bytes) -> char* {
        char* r = p;
        p += (bytes + 255) & ~(size_t)255;
        return r;
    };
    unsigned short* xb      = (unsigned short*)alloc((size_t)T_NUM * H_DIM * 2);
    unsigned short* actS    = (unsigned short*)alloc((size_t)T_NUM * IS_DIM * 2);
    unsigned short* actR    = (unsigned short*)alloc((size_t)T_NUM * TOPK * I_DIM * 2);
    int*   topk_idx  = (int*)alloc((size_t)T_NUM * TOPK * 4);
    float* topk_w    = (float*)alloc((size_t)T_NUM * TOPK * 4);
    int*   counts    = (int*)alloc(E_NUM * 4);
    int*   offsets   = (int*)alloc(E_NUM * 4);
    int*   cursors   = (int*)alloc(E_NUM * 4);
    int*   rowToken  = (int*)alloc((size_t)T_NUM * TOPK * 4);
    float* rowWeight = (float*)alloc((size_t)T_NUM * TOPK * 4);

    init_counts_kernel<<<1, 32, 0, stream>>>(counts);
    cvt_bf16_kernel<<<(T_NUM * H_DIM / 4 + 255) / 256, 256, 0, stream>>>(
        x, xb, T_NUM * H_DIM);
    router_kernel<<<T_NUM, 32, 0, stream>>>(x, gate_w, topk_idx, topk_w, counts);
    scan_kernel<<<1, 32, 0, stream>>>(counts, offsets, cursors);
    scatter_kernel<<<(T_NUM * TOPK + 255) / 256, 256, 0, stream>>>(
        topk_idx, topk_w, cursors, rowToken, rowWeight);

    // shared expert: gate_up + SwiGLU -> actS [T, IS] bf16
    gemm_wmma_kernel<0, false><<<dim3(IS_DIM / 64, T_NUM / 128, 1), 256, 0, stream>>>(
        xb, H_DIM, ws1, 0LL, H_DIM, IS_DIM, actS, IS_DIM, IS_DIM, H_DIM,
        nullptr, nullptr, T_NUM, nullptr, nullptr);
    // shared expert: down -> out [T, H] f32 (plain store initializes d_out)
    gemm_wmma_kernel<1, false><<<dim3(H_DIM / 64, T_NUM / 128, 1), 256, 0, stream>>>(
        actS, IS_DIM, ws2, 0LL, IS_DIM, 0, out, H_DIM, H_DIM, IS_DIM,
        nullptr, nullptr, T_NUM, nullptr, nullptr);
    // routed experts: gathered gate_up + SwiGLU -> actR (compacted rows)
    gemm_wmma_kernel<0, true><<<dim3(I_DIM / 64, T_NUM / 128, E_NUM), 256, 0, stream>>>(
        xb, H_DIM, w1, (long long)(2 * I_DIM) * H_DIM, H_DIM, I_DIM,
        actR, I_DIM, I_DIM, H_DIM,
        counts, offsets, 0, rowToken, rowWeight);
    // routed experts: down, scale by routing weight, atomic accumulate into out
    gemm_wmma_kernel<2, false><<<dim3(H_DIM / 64, T_NUM / 128, E_NUM), 256, 0, stream>>>(
        actR, I_DIM, w2, (long long)H_DIM * I_DIM, I_DIM, 0,
        out, H_DIM, H_DIM, I_DIM,
        counts, offsets, 0, rowToken, rowWeight);
}
